// RNNWavefunction_14585708937396
// MI455X (gfx1250) — matrix operand order repaired and checked by reading, not compiled, not run on hardware
//
#include <hip/hip_runtime.h>

// ---------------------------------------------------------------------------
// RNN wavefunction (2-layer GRU, H=256, 64 steps) for MI455X / gfx1250.
// Persistent workgroup per 128-sample batch tile; hidden states live in LDS
// for the whole sequence; all three per-step GEMMs run on
// v_wmma_f32_16x16x32_bf16 (bf16 operands, fp32 accumulation).
// ---------------------------------------------------------------------------

typedef unsigned short ushort_t;
typedef __attribute__((ext_vector_type(16))) __bf16        v16bf;
typedef __attribute__((ext_vector_type(8)))  float         v8f;
typedef __attribute__((ext_vector_type(4)))  unsigned int  u32x4;

namespace rnnwf {

constexpr int kB    = 32768;   // batch
constexpr int kSORB = 64;      // time steps
constexpr int kH    = 256;     // hidden
constexpr int kH3   = 768;     // 3*H (gates r,z,n)
constexpr int kTile = 128;     // batch rows per workgroup
constexpr int kHStr = kH + 8;  // padded LDS row stride (bf16 elems) -> bank spread
constexpr int kHBuf = kTile * kHStr;       // ushorts per h buffer
constexpr int kWElems = kH3 * kH;          // 196608 elems per weight matrix
constexpr size_t kWBytes = size_t(kWElems) * 2;

// workspace layout (bytes)
constexpr size_t kWsWhh0   = 0;
constexpr size_t kWsWih1   = kWBytes;
constexpr size_t kWsWhh1   = 2 * kWBytes;
constexpr size_t kWsColsel = 3 * kWBytes;  // 2*768 f32

// dynamic LDS layout (bytes)
constexpr size_t kOffColsel = size_t(4) * kHBuf * 2;  // after 4 h buffers
constexpr size_t kOffBih0   = kOffColsel + size_t(2) * kH3 * 4;
constexpr size_t kOffBhh0   = kOffBih0 + size_t(kH3) * 4;
constexpr size_t kOffBih1   = kOffBhh0 + size_t(kH3) * 4;
constexpr size_t kOffBhh1   = kOffBih1 + size_t(kH3) * 4;
constexpr size_t kOffWamp   = kOffBhh1 + size_t(kH3) * 4;
constexpr size_t kOffBamp   = kOffWamp + size_t(2) * kH * 4;
constexpr size_t kOffBits   = kOffBamp + 2 * 4;
constexpr size_t kSmemBytes = kOffBits + size_t(kTile) * 4;   // ~291 KB < 320 KB

} // namespace rnnwf

using namespace rnnwf;

// ------------------------------- helpers -----------------------------------

__device__ __forceinline__ ushort_t f2bf(float f) {
  unsigned u = __float_as_uint(f);
  unsigned r = ((u >> 16) & 1u) + 0x7fffu;   // round-to-nearest-even
  return (ushort_t)((u + r) >> 16);
}
__device__ __forceinline__ float bf2f(ushort_t b) {
  return __uint_as_float(((unsigned)b) << 16);
}
__device__ __forceinline__ float sigm_f(float x) {
  return 1.0f / (1.0f + __expf(-x));
}
__device__ __forceinline__ float tanh_f(float x) {
  x = fminf(15.0f, fmaxf(-15.0f, x));
  float e = __expf(-2.0f * x);
  return (1.0f - e) / (1.0f + e);
}

union FragBits { u32x4 q[2]; v16bf v; };

// A fragment (16x32 bf16) from row-major LDS tile (contiguous K, matches B):
// lane<16: row=lane,    K = k+0..15
// lane>=16:row=lane-16, K = k+16..31
__device__ __forceinline__ v16bf load_a_lds(const ushort_t* base, int m0,
                                            int kbase, int lane) {
  int row = m0 + (lane & 15);
  int k0  = kbase + ((lane >> 4) << 4);
  const ushort_t* p = base + row * kHStr + k0;
  FragBits f;
  f.q[0] = *(const u32x4*)(p);        // K k0..k0+7
  f.q[1] = *(const u32x4*)(p + 8);    // K k0+8..k0+15
  return f.v;
}

// B fragment (32x16 bf16). gh = h * W^T  =>  B[k][n] = W[n][k], so the
// row-major bf16 weight copy is already fragment-ordered:
// lane<16: col n=lane, K=kbase..kbase+15 ; lane>=16: K=kbase+16..kbase+31.
__device__ __forceinline__ v16bf load_b_w(const ushort_t* w, int n0,
                                          int kbase, int lane) {
  int n  = n0 + (lane & 15);
  int k0 = kbase + ((lane >> 4) << 4);
  const ushort_t* p = w + n * kH + k0;
  FragBits f;
  f.q[0] = *(const u32x4*)(p);
  f.q[1] = *(const u32x4*)(p + 8);
  return f.v;
}

__device__ __forceinline__ v8f wmma_bf16(v16bf a, v16bf b, v8f c) {
  return __builtin_amdgcn_wmma_f32_16x16x32_bf16(
      false, a, false, b, (short)0, c, false, false);
}

// One GRU layer for this wave's 16-row M tile. Gate columns (c, c+H, c+2H)
// are computed together so gate math fuses on the WMMA C-fragments.
template <bool L1>
__device__ __forceinline__ void gru_layer(
    const ushort_t* __restrict__ hc,     // LDS: h_{t-1} of this layer (bf16)
    const ushort_t* __restrict__ gi_in,  // LDS: layer input (L1 only)
    ushort_t* __restrict__ hn,           // LDS: h_t output (bf16)
    const ushort_t* __restrict__ whh,    // global bf16 (3H x H) row-major
    const ushort_t* __restrict__ wih,    // global bf16 (3H x H) (L1 only)
    const float* __restrict__ bhh,       // LDS f32 [3H]
    const float* __restrict__ bih,       // LDS f32 [3H]
    const float* __restrict__ colsel,    // LDS f32 [2][3H] (L0 only)
    const int* __restrict__ bits,        // LDS int [kTile] (L0, t>=0)
    int t, int m0, int lane) {

  const int N   = lane & 15;
  const int mhi = (lane >> 4) << 3;

  for (int c = 0; c < kH / 16; ++c) {
    // prefetch the three gate row-blocks of W_hh (and W_ih for L1)
    __builtin_prefetch(whh + (c * 16 + N) * kH, 0, 1);
    __builtin_prefetch(whh + ((kH + c * 16) + N) * kH, 0, 1);
    __builtin_prefetch(whh + ((2 * kH + c * 16) + N) * kH, 0, 1);
    if (L1) {
      __builtin_prefetch(wih + (c * 16 + N) * kH, 0, 1);
      __builtin_prefetch(wih + ((kH + c * 16) + N) * kH, 0, 1);
      __builtin_prefetch(wih + ((2 * kH + c * 16) + N) * kH, 0, 1);
    }

    v8f ar = {}, az = {}, an = {};   // gh accumulators (r,z,n)
    v8f gr = {}, gz = {}, gn = {};   // gi accumulators (L1)

#pragma unroll
    for (int k = 0; k < kH; k += 32) {
      v16bf a = load_a_lds(hc, m0, k, lane);
      ar = wmma_bf16(a, load_b_w(whh, c * 16, k, lane), ar);
      az = wmma_bf16(a, load_b_w(whh, kH + c * 16, k, lane), az);
      an = wmma_bf16(a, load_b_w(whh, 2 * kH + c * 16, k, lane), an);
      if (L1) {
        v16bf ai = load_a_lds(gi_in, m0, k, lane);
        gr = wmma_bf16(ai, load_b_w(wih, c * 16, k, lane), gr);
        gz = wmma_bf16(ai, load_b_w(wih, kH + c * 16, k, lane), gz);
        gn = wmma_bf16(ai, load_b_w(wih, 2 * kH + c * 16, k, lane), gn);
      }
    }

    const int col = c * 16 + N;
#pragma unroll
    for (int v = 0; v < 8; ++v) {
      int rl = m0 + v + mhi;         // local batch row (C layout: M=v+8*hi)
      float ghr = ar[v] + bhh[col];
      float ghz = az[v] + bhh[kH + col];
      float ghn = an[v] + bhh[2 * kH + col];
      float gir, giz, gin;
      if (L1) {
        gir = gr[v] + bih[col];
        giz = gz[v] + bih[kH + col];
        gin = gn[v] + bih[2 * kH + col];
      } else if (t < 0) {            // priming step: x = 0 -> gi = b_ih0
        gir = bih[col];
        giz = bih[kH + col];
        gin = bih[2 * kH + col];
      } else {                       // one-hot input -> column select
        const float* cs = colsel + bits[rl] * kH3;
        gir = cs[col];
        giz = cs[kH + col];
        gin = cs[2 * kH + col];
      }
      float r  = sigm_f(gir + ghr);
      float z  = sigm_f(giz + ghz);
      float nn = tanh_f(gin + r * ghn);
      float hp = bf2f(hc[rl * kHStr + col]);
      float hv = (1.0f - z) * nn + z * hp;
      hn[rl * kHStr + col] = f2bf(hv);
    }
  }
}

// -------------------------- weight pack kernel -----------------------------
// f32 -> bf16 row-major copies of w_hh0, w_ih1, w_hh1; colsel[b][j] =
// w_ih0[j][b] + b_ih0[j] (the entire layer-0 input GEMM for one-hot inputs).
__global__ void rnnwf_pack_kernel(const float* __restrict__ wih0,
                                  const float* __restrict__ whh0,
                                  const float* __restrict__ wih1,
                                  const float* __restrict__ whh1,
                                  const float* __restrict__ bih0,
                                  ushort_t* __restrict__ whh0_b,
                                  ushort_t* __restrict__ wih1_b,
                                  ushort_t* __restrict__ whh1_b,
                                  float* __restrict__ colsel) {
  int i = blockIdx.x * blockDim.x + threadIdx.x;
  if (i < kWElems) {
    whh0_b[i] = f2bf(whh0[i]);
  } else if (i < 2 * kWElems) {
    int j = i - kWElems;
    wih1_b[j] = f2bf(wih1[j]);
  } else if (i < 3 * kWElems) {
    int j = i - 2 * kWElems;
    whh1_b[j] = f2bf(whh1[j]);
  } else {
    int j = i - 3 * kWElems;
    if (j < 2 * kH3) {
      int b = j / kH3, cidx = j % kH3;
      colsel[j] = wih0[cidx * 2 + b] + bih0[cidx];
    }
  }
}

// ------------------------------ main kernel --------------------------------
__global__ __launch_bounds__(256, 1)
void rnnwf_main_kernel(const int* __restrict__ x,
                       const ushort_t* __restrict__ whh0,
                       const ushort_t* __restrict__ wih1,
                       const ushort_t* __restrict__ whh1,
                       const float* __restrict__ colsel_g,
                       const float* __restrict__ bih0_g,
                       const float* __restrict__ bhh0_g,
                       const float* __restrict__ bih1_g,
                       const float* __restrict__ bhh1_g,
                       const float* __restrict__ wamp_g,
                       const float* __restrict__ bamp_g,
                       float* __restrict__ out) {
  extern __shared__ char smem[];
  ushort_t* hb   = (ushort_t*)smem;                 // 4 x kHBuf bf16 h tiles
  float* colsel  = (float*)(smem + kOffColsel);
  float* bih0    = (float*)(smem + kOffBih0);
  float* bhh0    = (float*)(smem + kOffBhh0);
  float* bih1    = (float*)(smem + kOffBih1);
  float* bhh1    = (float*)(smem + kOffBhh1);
  float* wamp    = (float*)(smem + kOffWamp);
  float* bamp    = (float*)(smem + kOffBamp);
  int*   bits    = (int*)(smem + kOffBits);

  const int tid     = threadIdx.x;
  const int lane    = tid & 31;
  const int m0      = (tid >> 5) * 16;      // wave -> 16-row M tile
  const int rowbase = blockIdx.x * kTile;

  // stage constants + zero hidden state
  for (int i = tid; i < 4 * kHBuf; i += 256) hb[i] = 0;
  for (int i = tid; i < 2 * kH3; i += 256) colsel[i] = colsel_g[i];
  for (int i = tid; i < kH3; i += 256) {
    bih0[i] = bih0_g[i];
    bhh0[i] = bhh0_g[i];
    bih1[i] = bih1_g[i];
    bhh1[i] = bhh1_g[i];
  }
  for (int i = tid; i < 2 * kH; i += 256) wamp[i] = wamp_g[i];
  if (tid < 2) bamp[tid] = bamp_g[tid];
  __syncthreads();

  float logacc = 0.0f;

  for (int it = 0; it <= kSORB; ++it) {   // it==0 is the zero-input priming step
    const int t   = it - 1;
    const int cur = it & 1;
    const int nxt = cur ^ 1;
    ushort_t* h0c = hb + (0 + cur) * kHBuf;
    ushort_t* h0n = hb + (0 + nxt) * kHBuf;
    ushort_t* h1c = hb + (2 + cur) * kHBuf;
    ushort_t* h1n = hb + (2 + nxt) * kHBuf;

    if (t >= 0 && tid < kTile) bits[tid] = x[(rowbase + tid) * kSORB + t];
    __syncthreads();

    gru_layer<false>(h0c, nullptr, h0n, whh0, nullptr,
                     bhh0, bih0, colsel, bits, t, m0, lane);
    __syncthreads();

    gru_layer<true>(h1c, h0n, h1n, whh1, wih1,
                    bhh1, bih1, nullptr, bits, t, m0, lane);
    __syncthreads();

    if (t >= 0 && tid < kTile) {
      // amp logits for this row; p = sigmoid(l_bit - l_other)
      const ushort_t* hr = h1n + tid * kHStr;
      float l0 = bamp[0], l1 = bamp[1];
#pragma unroll 8
      for (int k = 0; k < kH; ++k) {
        float hv = bf2f(hr[k]);
        l0 = fmaf(hv, wamp[k], l0);
        l1 = fmaf(hv, wamp[kH + k], l1);
      }
      int b   = bits[tid];
      float d = b ? (l0 - l1) : (l1 - l0);       // other - chosen
      float p = 1.0f / (1.0f + __expf(d));
      logacc += __logf(p);
    }
  }

  if (tid < kTile) out[rowbase + tid] = __expf(0.5f * logacc);
}

// ------------------------------ launcher -----------------------------------
extern "C" void kernel_launch(void* const* d_in, const int* in_sizes, int n_in,
                              void* d_out, int out_size, void* d_ws,
                              size_t ws_size, hipStream_t stream) {
  const int*   x     = (const int*)d_in[0];
  const float* wih0  = (const float*)d_in[1];
  const float* whh0f = (const float*)d_in[2];
  const float* bih0  = (const float*)d_in[3];
  const float* bhh0  = (const float*)d_in[4];
  const float* wih1f = (const float*)d_in[5];
  const float* whh1f = (const float*)d_in[6];
  const float* bih1  = (const float*)d_in[7];
  const float* bhh1  = (const float*)d_in[8];
  const float* wamp  = (const float*)d_in[9];
  const float* bamp  = (const float*)d_in[10];

  char* ws = (char*)d_ws;
  ushort_t* whh0_b = (ushort_t*)(ws + kWsWhh0);
  ushort_t* wih1_b = (ushort_t*)(ws + kWsWih1);
  ushort_t* whh1_b = (ushort_t*)(ws + kWsWhh1);
  float*    colsel = (float*)(ws + kWsColsel);

  const int npack = 3 * kWElems + 2 * kH3;
  rnnwf_pack_kernel<<<(npack + 255) / 256, 256, 0, stream>>>(
      wih0, whh0f, wih1f, whh1f, bih0, whh0_b, wih1_b, whh1_b, colsel);

  rnnwf_main_kernel<<<kB / kTile, 256, kSmemBytes, stream>>>(
      x, whh0_b, wih1_b, whh1_b, colsel, bih0, bhh0, bih1, bhh1, wamp, bamp,
      (float*)d_out);
}
